// AttentionHead_88562225643755
// MI455X (gfx1250) — matrix operand order repaired and checked by reading, not compiled, not run on hardware
//
#include <hip/hip_runtime.h>

#define D_MODEL 512
#define DK      64
#define BATCH   8
#define SEQ     2048
#define NEG_INF -1.0e9f

typedef __attribute__((ext_vector_type(16))) _Float16 v16h;
typedef __attribute__((ext_vector_type(8)))  _Float16 v8h;
typedef __attribute__((ext_vector_type(8)))  float    v8f;
typedef __attribute__((ext_vector_type(4)))  float    v4f;
typedef __attribute__((ext_vector_type(4)))  unsigned int v4u;
typedef __attribute__((ext_vector_type(8)))  int      v8i;
typedef __attribute__((ext_vector_type(4)))  int      v4i;

#if defined(__has_builtin)
#if __has_builtin(__builtin_amdgcn_tensor_load_to_lds) && \
    __has_builtin(__builtin_amdgcn_s_wait_tensorcnt)
#define USE_TDM 1
#endif
#endif
#ifndef USE_TDM
#define USE_TDM 0
#endif

__device__ __forceinline__ v8f wmma_f32_f16(v16h a, v16h b, v8f c) {
  return __builtin_amdgcn_wmma_f32_16x16x32_f16(
      false, a, false, b, (short)0, c, false, false);
}
__device__ __forceinline__ v16h cat16(v8h lo, v8h hi) {
  return __builtin_shufflevector(lo, hi, 0, 1, 2, 3, 4, 5, 6, 7,
                                         8, 9, 10, 11, 12, 13, 14, 15);
}
__device__ __forceinline__ v8h ld8h(const _Float16* p) { return *(const v8h*)p; }

#if USE_TDM
// Issue a TDM 2D tile load (f16 elements) into LDS.  D# per CDNA5 ISA §8.
// amdgpu-toolchain (clang-23) 6-arg form:
//   (uint32x4 g0, int32x8 g1, int32x4 g2, int32x4 g3, int32x8, i32 cpol)
__device__ __forceinline__ void tdm_load_2d(
    const _Float16* gaddr, unsigned int lds_off,
    unsigned int tile_d0, unsigned int tile_d1,
    unsigned long long stride0, unsigned int tdim0, unsigned int tdim1)
{
  const unsigned long long ga = (unsigned long long)(uintptr_t)gaddr;
  v4u g0;
  g0[0] = 1u;                                           // count=1, user mode
  g0[1] = lds_off;                                      // lds_addr (bytes)
  g0[2] = (unsigned int)ga;                             // global_addr[31:0]
  g0[3] = (unsigned int)((ga >> 32) & 0x01FFFFFFu)      // global_addr[56:32]
          | (2u << 30);                                 // type = 2 ("image")
  v8i g1;
  g1[0] = (int)(1u << 16);                              // data_size=1 -> 2B
  g1[1] = (int)((tdim0 & 0xFFFFu) << 16);               // tensor_dim0[15:0]
  g1[2] = (int)((tdim0 >> 16) | ((tdim1 & 0xFFFFu) << 16));
  g1[3] = (int)((tdim1 >> 16) | (tile_d0 << 16));       // tile_dim0
  g1[4] = (int)(tile_d1 & 0xFFFFu);                     // tile_dim1, dim2=0
  g1[5] = (int)(stride0 & 0xFFFFFFFFull);               // dim0_stride[31:0]
  g1[6] = (int)((stride0 >> 32) & 0xFFFFull);           // [47:32], dim1_stride=0
  g1[7] = 0;
  const v4i z4 = {0, 0, 0, 0};                          // groups 2/3: <=2D tensor
  const v8i z8 = {0, 0, 0, 0, 0, 0, 0, 0};
  __builtin_amdgcn_tensor_load_to_lds(g0, g1, z4, z4, z8, 0);
}
__device__ __forceinline__ unsigned int lds_off(const void* p) {
  return (unsigned int)(uintptr_t)p;   // low 32 bits of LDS-aperture address
}
#endif

// ---------------------------------------------------------------------------
// Phase 0: transpose + downconvert weights: Wt[p][n][k] f16, p in {q,k,v}.
// ---------------------------------------------------------------------------
__global__ __launch_bounds__(256) void wt_kernel(
    const float* __restrict__ Wq, const float* __restrict__ Wk,
    const float* __restrict__ Wv, _Float16* __restrict__ Wt)
{
  const int i = blockIdx.x * 256 + threadIdx.x;   // over 3*64*512
  const int p = i >> 15;
  const int n = (i >> 9) & 63;
  const int k = i & 511;
  const float* W = (p == 0) ? Wq : ((p == 1) ? Wk : Wv);
  Wt[i] = (_Float16)W[(size_t)k * DK + n];
}

// ---------------------------------------------------------------------------
// Phase 1: QKV projection.  One wave computes a 16-row tile of Q, K and V.
// ---------------------------------------------------------------------------
__global__ __launch_bounds__(256) void qkv_proj_kernel(
    const float* __restrict__ q_src, const float* __restrict__ k_src,
    const float* __restrict__ v_src,
    const _Float16* __restrict__ Wt,
    const float* __restrict__ bq, const float* __restrict__ bk,
    const float* __restrict__ bv,
    _Float16* __restrict__ Qh, _Float16* __restrict__ Kh,
    _Float16* __restrict__ Vt)
{
  const int lane = threadIdx.x & 31;
  const int wave = threadIdx.x >> 5;
  const int tile = blockIdx.x * 8 + wave;   // 16-row tile over B*S
  const int row0 = tile * 16;
  const int h    = lane >> 4;
  const int l16  = lane & 15;

  const float* srcs[3] = {q_src, k_src, v_src};
  const float* bs[3]   = {bq, bk, bv};
  const float  scl[3]  = {0.125f, 1.0f, 1.0f};   // fold 1/sqrt(64) into Q

#pragma unroll
  for (int p = 0; p < 3; ++p) {
    const float* src = srcs[p];
    const _Float16* Wp = Wt + (size_t)p * DK * D_MODEL;
    v8f acc[4] = {v8f{}, v8f{}, v8f{}, v8f{}};
    for (int k0 = 0; k0 < D_MODEL; k0 += 32) {
      const float* arow = src + (size_t)(row0 + l16) * D_MODEL + k0 + 8 * h;
      const v4f a0 = *(const v4f*)(arow);
      const v4f a1 = *(const v4f*)(arow + 4);
      const v4f a2 = *(const v4f*)(arow + 16);
      const v4f a3 = *(const v4f*)(arow + 20);
      v16h a;
#pragma unroll
      for (int e = 0; e < 4; ++e) {
        a[e]      = (_Float16)a0[e];
        a[4 + e]  = (_Float16)a1[e];
        a[8 + e]  = (_Float16)a2[e];
        a[12 + e] = (_Float16)a3[e];
      }
#pragma unroll
      for (int t = 0; t < 4; ++t) {
        const _Float16* wrow =
            Wp + (size_t)(16 * t + l16) * D_MODEL + k0 + 8 * h;
        const v16h bm = cat16(ld8h(wrow), ld8h(wrow + 16));
        acc[t] = wmma_f32_f16(a, bm, acc[t]);
      }
    }
    const float s = scl[p];
    if (p < 2) {
      _Float16* dst = (p == 0) ? Qh : Kh;      // row-major [B*S][64]
#pragma unroll
      for (int t = 0; t < 4; ++t) {
        const int n = 16 * t + l16;
        const float bias = bs[p][n];
#pragma unroll
        for (int r = 0; r < 8; ++r) {
          const int m = row0 + r + 8 * h;
          dst[(size_t)m * DK + n] = (_Float16)((acc[t][r] + bias) * s);
        }
      }
    } else {
      // V stored transposed: Vt[b][n][s]; contiguous 8-row run per lane.
      const int b  = row0 / SEQ;
      const int s0 = row0 % SEQ;
      _Float16* dstb = Vt + (size_t)b * DK * SEQ;
#pragma unroll
      for (int t = 0; t < 4; ++t) {
        const int n = 16 * t + l16;
        const float bias = bs[2][n];
        v8h ov;
#pragma unroll
        for (int r = 0; r < 8; ++r) ov[r] = (_Float16)(acc[t][r] + bias);
        *(v8h*)(dstb + (size_t)n * SEQ + s0 + 8 * h) = ov;
      }
    }
  }
}

// ---------------------------------------------------------------------------
// Phase 2: causal flash attention.  One block = 64 query rows = 4 waves
// sharing TDM-double-buffered K/V tiles in LDS (32 keys per step).
// ---------------------------------------------------------------------------
__global__ __launch_bounds__(128) void flash_attn_kernel(
    const _Float16* __restrict__ Qh, const _Float16* __restrict__ Kh,
    const _Float16* __restrict__ Vt, float* __restrict__ out)
{
  __shared__ _Float16 ks[2][32 * DK];     // K tile:  [key 0..31][d 0..63]
  __shared__ _Float16 vs[2][DK * 32];     // Vt tile: [d 0..63][key 0..31]
  __shared__ _Float16 p_lds[4][16 * 32];  // per-wave P staging

  const int lane = threadIdx.x & 31;
  const int wave = threadIdx.x >> 5;
  const int blk  = blockIdx.x;            // 256 blocks over B*S/64
  const int b    = blk / (SEQ / 64);
  const int q0   = (blk % (SEQ / 64)) * 64;
  const int q0w  = q0 + 16 * wave;        // this wave's 16-row query tile
  const int h    = lane >> 4;
  const int l16  = lane & 15;

  const _Float16* Qb  = Qh + (size_t)b * SEQ * DK;
  const _Float16* Kb  = Kh + (size_t)b * SEQ * DK;
  const _Float16* Vtb = Vt + (size_t)b * DK * SEQ;

  // Q tile in A layout (two 32-wide d chunks, contiguous 8-f16 runs)
  v16h qa[2];
  {
    const _Float16* qrow = Qb + (size_t)(q0w + l16) * DK + 16 * h;
#pragma unroll
    for (int c = 0; c < 2; ++c)
      qa[c] = cat16(ld8h(qrow + 32 * c), ld8h(qrow + 32 * c + 8));
  }

  float m_st[8], l_st[8];
  v8f o_acc[4] = {v8f{}, v8f{}, v8f{}, v8f{}};
#pragma unroll
  for (int r = 0; r < 8; ++r) { m_st[r] = -3.0e38f; l_st[r] = 0.0f; }

  const int nit = q0 / 32 + 2;            // covers keys 0 .. q0+63

#if USE_TDM
  if (wave == 0) {                        // TDM descriptors are wave-uniform
    tdm_load_2d(Kb,  lds_off(&ks[0][0]), DK, 32, DK,  DK, 1u << 20);
    tdm_load_2d(Vtb, lds_off(&vs[0][0]), 32, DK, SEQ, 1u << 20, 1u << 20);
  }
#endif

  for (int i = 0; i < nit; ++i) {
    const int kb = 32 * i;
    const int p  = i & 1;
#if USE_TDM
    if (wave == 0) __builtin_amdgcn_s_wait_tensorcnt((short)0);
    __syncthreads();                      // tile i visible to all waves
    if (wave == 0 && i + 1 < nit) {       // prefetch tile i+1 into other buffer
      const int kn = kb + 32;
      tdm_load_2d(Kb + (size_t)kn * DK, lds_off(&ks[p ^ 1][0]),
                  DK, 32, DK, DK, 1u << 20);
      tdm_load_2d(Vtb + kn, lds_off(&vs[p ^ 1][0]),
                  32, DK, SEQ, 1u << 20, 1u << 20);
    }
#else
    __syncthreads();                      // all reads of buffer 0 done
    {   // cooperative copy, 128 threads x 2 chunks of 16B each for K and V
      const int tid = threadIdx.x;
#pragma unroll
      for (int j = 0; j < 2; ++j) {
        const int u = tid + 128 * j;      // 256 units of 8 f16
        *(v8h*)&ks[0][u * 8] = ld8h(Kb + (size_t)kb * DK + u * 8);
        const int d = u >> 2, x8 = (u & 3) * 8;
        *(v8h*)&vs[0][d * 32 + x8] = ld8h(Vtb + (size_t)d * SEQ + kb + x8);
      }
    }
    __syncthreads();
#endif
    const _Float16* kst = &ks[USE_TDM ? p : 0][0];
    const _Float16* vst = &vs[USE_TDM ? p : 0][0];

    // ---- scores: S1 = Q*K[kb..kb+16)^T ; S2 = Q*K[kb+16..kb+32)^T
    v8f s1 = {}, s2 = {};
#pragma unroll
    for (int c = 0; c < 2; ++c) {
      const _Float16* kr1 = kst + (size_t)l16 * DK + 32 * c + 16 * h;
      const _Float16* kr2 = kr1 + (size_t)16 * DK;
      const v16h b1 = cat16(ld8h(kr1), ld8h(kr1 + 8));
      const v16h b2 = cat16(ld8h(kr2), ld8h(kr2 + 8));
      s1 = wmma_f32_f16(qa[c], b1, s1);
      s2 = wmma_f32_f16(qa[c], b2, s2);
    }
    // ---- causal mask (per-lane selects; EXEC stays all-ones)
    const int k1 = kb + l16, k2 = kb + 16 + l16;
#pragma unroll
    for (int r = 0; r < 8; ++r) {
      const int q = q0w + r + 8 * h;
      if (k1 > q) s1[r] = NEG_INF;
      if (k2 > q) s2[r] = NEG_INF;
    }
    // ---- online softmax (row lives in one 16-lane half: xor 1/2/4/8)
    float alpha[8];
#pragma unroll
    for (int r = 0; r < 8; ++r) {
      float mx = fmaxf(s1[r], s2[r]);
#pragma unroll
      for (int m = 1; m < 16; m <<= 1) mx = fmaxf(mx, __shfl_xor(mx, m, 32));
      const float mnew = fmaxf(m_st[r], mx);
      alpha[r] = __expf(m_st[r] - mnew);
      m_st[r]  = mnew;
      const float p1 = __expf(s1[r] - mnew);
      const float p2 = __expf(s2[r] - mnew);
      s1[r] = p1; s2[r] = p2;
      float rs = p1 + p2;
#pragma unroll
      for (int m = 1; m < 16; m <<= 1) rs += __shfl_xor(rs, m, 32);
      l_st[r] = l_st[r] * alpha[r] + rs;
    }
#pragma unroll
    for (int t = 0; t < 4; ++t)
#pragma unroll
      for (int r = 0; r < 8; ++r) o_acc[t][r] *= alpha[r];

    // ---- P: C layout -> row-major LDS -> A layout (DS in-order per wave)
    _Float16* pl = p_lds[wave];
#pragma unroll
    for (int r = 0; r < 8; ++r) {
      pl[(r + 8 * h) * 32 + l16]      = (_Float16)s1[r];
      pl[(r + 8 * h) * 32 + 16 + l16] = (_Float16)s2[r];
    }
    const _Float16* prow = pl + l16 * 32 + 8 * h;
    const v16h pa = cat16(*(const v8h*)prow, *(const v8h*)(prow + 16));

    // ---- O += P(16x32) * V(32x64): B from Vt tile, contiguous along keys
#pragma unroll
    for (int t = 0; t < 4; ++t) {
      const _Float16* vrow = vst + (size_t)(16 * t + l16) * 32 + 8 * h;
      const v16h bv_ = cat16(ld8h(vrow), ld8h(vrow + 16));
      o_acc[t] = wmma_f32_f16(pa, bv_, o_acc[t]);
    }
  }

  // ---- normalize, store fp32 output
  float* outb = out + (size_t)b * SEQ * DK;
#pragma unroll
  for (int t = 0; t < 4; ++t) {
    const int n = 16 * t + l16;
#pragma unroll
    for (int r = 0; r < 8; ++r) {
      const int q = q0w + r + 8 * h;
      outb[(size_t)q * DK + n] = o_acc[t][r] / l_st[r];
    }
  }
}

extern "C" void kernel_launch(void* const* d_in, const int* in_sizes, int n_in,
                              void* d_out, int out_size, void* d_ws, size_t ws_size,
                              hipStream_t stream) {
  const float* q_src = (const float*)d_in[0];
  const float* k_src = (const float*)d_in[1];
  const float* v_src = (const float*)d_in[2];
  // d_in[3] = mask: unused, causal structure computed analytically
  const float* Wq = (const float*)d_in[4];
  const float* bq = (const float*)d_in[5];
  const float* Wk = (const float*)d_in[6];
  const float* bk = (const float*)d_in[7];
  const float* Wv = (const float*)d_in[8];
  const float* bv = (const float*)d_in[9];

  _Float16* Qh = (_Float16*)d_ws;                    // [B*S][64] f16, pre-scaled
  _Float16* Kh = Qh + (size_t)BATCH * SEQ * DK;      // [B*S][64] f16
  _Float16* Vt = Kh + (size_t)BATCH * SEQ * DK;      // [B][64][S] f16 transposed
  _Float16* Wt = Vt + (size_t)BATCH * SEQ * DK;      // [3][64][512] f16

  wt_kernel<<<dim3(3 * DK * D_MODEL / 256), dim3(256), 0, stream>>>(
      Wq, Wk, Wv, Wt);
  qkv_proj_kernel<<<dim3(128), dim3(256), 0, stream>>>(
      q_src, k_src, v_src, Wt, bq, bk, bv, Qh, Kh, Vt);
  flash_attn_kernel<<<dim3(BATCH * SEQ / 64), dim3(128), 0, stream>>>(
      Qh, Kh, Vt, (float*)d_out);
}
